// MyModel_61933428415948
// MI455X (gfx1250) — matrix operand order repaired and checked by reading, not compiled, not run on hardware
//
#include <hip/hip_runtime.h>
#include <hip/hip_bf16.h>
#include <math.h>

// ---------------------------------------------------------------------------
// LED encoder forward, MI455X (gfx1250): bf16 WMMA (v_wmma_f32_16x16x32_bf16)
// for all GEMMs and sliding-window attention; f32 VALU for LN/softmax.
// Weights converted to bf16 [N,K] per layer (L2-resident, 192MB), activations
// mirrored in bf16. Compute-bound workload (~465 GFLOP vs ~10us of HBM time).
// GEMM register-blocks N by 4 (16x64 per wave) so one A fragment feeds four
// WMMAs: 4 wmma per 10 b128 loads instead of 1 per 4.
// ---------------------------------------------------------------------------

typedef __attribute__((ext_vector_type(16))) __bf16 v16bf;
typedef __attribute__((ext_vector_type(8)))  float  v8f;

union Frag {
    uint32_t        u[8];
    unsigned short  h[16];
    v16bf           v;
};

__device__ __forceinline__ unsigned short f2bf(float f) {
    union { float f; uint32_t u; } c; c.f = f;
    uint32_t u = c.u;
    uint32_t r = u + 0x7fffu + ((u >> 16) & 1u);   // round-to-nearest-even
    return (unsigned short)(r >> 16);
}

// A-pattern dword index (16-bit A matrix 16x32): lanes 0-15 K=0..15,
// lanes 16-31 K=16..31; element pair j <-> one dword.
__device__ __forceinline__ int aidx(int j, int half) {
    return j + (half ? 8 : 0);
}
// B-pattern dword index (32x16 B): lanes 0-15 K=0..15, lanes 16-31 K=16..31.
__device__ __forceinline__ int bidx(int j, int half) {
    return j + (half ? 8 : 0);
}

// ---------------------------------------------------------------------------
// GEMM: C[M,N] = act( alpha * (A[M,K](bf16) * Bt[N,K](bf16)^T + bias) )
// One wave per 16x64 strip (4 accumulators), 8 waves/block.
// K-loop: 1 A fragment + 4 B fragments -> 4 x 16x16x32 bf16 WMMA.
// ---------------------------------------------------------------------------
__global__ void gemm_bf16_wmma(const unsigned short* __restrict__ A,
                               const unsigned short* __restrict__ Bt,
                               const float* __restrict__ bias,
                               float* __restrict__ Cf,
                               unsigned short* __restrict__ Cb,
                               int M, int N, int K, float alpha, int act_gelu) {
    const int lane = threadIdx.x & 31;
    const int wv   = threadIdx.x >> 5;
    const int nl   = lane & 15;
    const int half = lane >> 4;

    const int tile = blockIdx.x * 8 + wv;
    const int ncg  = N >> 6;               // 64-wide column groups
    const int tm   = tile / ncg;
    const int tg   = tile - tm * ncg;
    if (tm * 16 >= M) return;

    const uint32_t* ap = (const uint32_t*)(A + (size_t)(tm * 16 + nl) * K);
    const uint32_t* bp[4];
#pragma unroll
    for (int t = 0; t < 4; ++t)
        bp[t] = (const uint32_t*)(Bt + (size_t)(tg * 64 + t * 16 + nl) * K);

    v8f c0 = {0.f,0.f,0.f,0.f,0.f,0.f,0.f,0.f};
    v8f c1 = c0, c2 = c0, c3 = c0;

    for (int k0 = 0; k0 < K; k0 += 32) {
        const int kd = k0 >> 1;
        Frag a;
#pragma unroll
        for (int j = 0; j < 8; ++j) a.u[j] = ap[kd + aidx(j, half)];

        Frag b0, b1, b2, b3;
#pragma unroll
        for (int j = 0; j < 8; ++j) {
            const int bi = kd + bidx(j, half);
            b0.u[j] = bp[0][bi];
            b1.u[j] = bp[1][bi];
            b2.u[j] = bp[2][bi];
            b3.u[j] = bp[3][bi];
        }
        c0 = __builtin_amdgcn_wmma_f32_16x16x32_bf16(false, a.v, false, b0.v,
                                                     (short)0, c0, false, false);
        c1 = __builtin_amdgcn_wmma_f32_16x16x32_bf16(false, a.v, false, b1.v,
                                                     (short)0, c1, false, false);
        c2 = __builtin_amdgcn_wmma_f32_16x16x32_bf16(false, a.v, false, b2.v,
                                                     (short)0, c2, false, false);
        c3 = __builtin_amdgcn_wmma_f32_16x16x32_bf16(false, a.v, false, b3.v,
                                                     (short)0, c3, false, false);
    }

    v8f acc[4] = {c0, c1, c2, c3};
#pragma unroll
    for (int t = 0; t < 4; ++t) {
        const int col = tg * 64 + t * 16 + nl;
#pragma unroll
        for (int r = 0; r < 8; ++r) {
            const int row = tm * 16 + r + 8 * half;
            float v = acc[t][r];
            if (bias) v += bias[col];
            v *= alpha;
            if (act_gelu) v = 0.5f * v * (1.f + erff(v * 0.70710678118654752f));
            const size_t o = (size_t)row * N + col;
            if (Cf) Cf[o] = v;
            if (Cb) Cb[o] = f2bf(v);
        }
    }
}

// ---------------------------------------------------------------------------
// Sliding-window attention. One block (4 waves, 128 thr) per (16-query tile,
// head). Band = [q0-512, q0+15+512] -> 1040 keys, padded to 1056 (33x32).
// pass1: QK^T row-max (recompute scheme, no f32 score LDS needed)
// pass2: recompute QK^T, exp, bf16 P -> LDS, row-sum
// pass3: P*V via WMMA, normalize at store, emit bf16 [S,dm]
// ---------------------------------------------------------------------------
#define ATT_BAND 1056
#define ATT_NT16 66
#define ATT_NT32 33

__device__ __forceinline__ v8f qk_tile(const Frag& aQ0, const Frag& aQ1,
                                       const unsigned short* __restrict__ Kb,
                                       int kc, int dm, int h64, int half) {
    const uint32_t* kp = (const uint32_t*)(Kb + (size_t)kc * dm + h64);
    Frag b0, b1;
#pragma unroll
    for (int j = 0; j < 8; ++j) {
        b0.u[j] = kp[bidx(j, half)];        // d = 0..31
        b1.u[j] = kp[16 + bidx(j, half)];   // d = 32..63
    }
    v8f c = {0.f, 0.f, 0.f, 0.f, 0.f, 0.f, 0.f, 0.f};
    c = __builtin_amdgcn_wmma_f32_16x16x32_bf16(false, aQ0.v, false, b0.v,
                                                (short)0, c, false, false);
    c = __builtin_amdgcn_wmma_f32_16x16x32_bf16(false, aQ1.v, false, b1.v,
                                                (short)0, c, false, false);
    return c;
}

__global__ void attn_window_wmma(const unsigned short* __restrict__ Qb,
                                 const unsigned short* __restrict__ Kb,
                                 const unsigned short* __restrict__ Vb,
                                 unsigned short* __restrict__ Ab,
                                 int S, int dm, int w) {
    const int qt   = blockIdx.x;
    const int head = blockIdx.y;
    const int q0   = qt * 16;
    const int h64  = head * 64;
    const int tid  = threadIdx.x;
    const int wv   = tid >> 5;
    const int lane = tid & 31;
    const int nl   = lane & 15;
    const int half = lane >> 4;
    const int kstart = q0 - w;

    __shared__ unsigned short sP[16 * ATT_BAND];       // exp(score - max), bf16
    __shared__ float partmax[4][16], partsum[4][16];
    __shared__ float rowmax[16], rowsum[16];

    // Q fragments (row = q0+nl), reused across all key tiles
    const uint32_t* qp = (const uint32_t*)(Qb + (size_t)(q0 + nl) * dm + h64);
    Frag aQ0, aQ1;
#pragma unroll
    for (int j = 0; j < 8; ++j) {
        aQ0.u[j] = qp[aidx(j, half)];
        aQ1.u[j] = qp[16 + aidx(j, half)];
    }

    // ---- pass 1: row max over the band ----
    float rmax[8];
#pragma unroll
    for (int r = 0; r < 8; ++r) rmax[r] = -3.0e38f;

    for (int kt = wv; kt < ATT_NT16; kt += 4) {
        const int key = kstart + kt * 16 + nl;
        const int kc  = key < 0 ? 0 : (key >= S ? S - 1 : key);
        v8f c = qk_tile(aQ0, aQ1, Kb, kc, dm, h64, half);
#pragma unroll
        for (int r = 0; r < 8; ++r) {
            const int q = q0 + r + 8 * half;
            const int d = key - q;
            const bool ok = (key >= 0) && (key < S) && (d <= w) && (-d <= w);
            const float sc = ok ? c[r] : -1.0e9f;
            rmax[r] = fmaxf(rmax[r], sc);
        }
    }
#pragma unroll
    for (int r = 0; r < 8; ++r) {
        float v = rmax[r];
        for (int off = 8; off >= 1; off >>= 1)
            v = fmaxf(v, __shfl_xor(v, off, 32));
        if (nl == 0) partmax[wv][r + 8 * half] = v;
    }
    __syncthreads();
    if (tid < 16) {
        float m = partmax[0][tid];
        for (int p = 1; p < 4; ++p) m = fmaxf(m, partmax[p][tid]);
        rowmax[tid] = m;
    }
    __syncthreads();

    // ---- pass 2: exp, store bf16 P, row sums ----
    float rsum[8];
#pragma unroll
    for (int r = 0; r < 8; ++r) rsum[r] = 0.f;

    for (int kt = wv; kt < ATT_NT16; kt += 4) {
        const int key = kstart + kt * 16 + nl;
        const int kc  = key < 0 ? 0 : (key >= S ? S - 1 : key);
        v8f c = qk_tile(aQ0, aQ1, Kb, kc, dm, h64, half);
#pragma unroll
        for (int r = 0; r < 8; ++r) {
            const int row = r + 8 * half;
            const int q = q0 + row;
            const int d = key - q;
            const bool ok = (key >= 0) && (key < S) && (d <= w) && (-d <= w);
            const float e = ok ? __expf(c[r] - rowmax[row]) : 0.f;
            sP[row * ATT_BAND + kt * 16 + nl] = f2bf(e);
            rsum[r] += e;
        }
    }
#pragma unroll
    for (int r = 0; r < 8; ++r) {
        float v = rsum[r];
        for (int off = 8; off >= 1; off >>= 1)
            v += __shfl_xor(v, off, 32);
        if (nl == 0) partsum[wv][r + 8 * half] = v;
    }
    __syncthreads();
    if (tid < 16) {
        float s = partsum[0][tid];
        for (int p = 1; p < 4; ++p) s += partsum[p][tid];
        rowsum[tid] = s;
    }
    __syncthreads();

    // ---- pass 3: out = (P @ V) / rowsum ; wave wv owns 16 head-dims ----
    const int dcol = h64 + wv * 16 + nl;
    v8f o = {0.f, 0.f, 0.f, 0.f, 0.f, 0.f, 0.f, 0.f};
    for (int kt = 0; kt < ATT_NT32; ++kt) {
        Frag a, b;
        const uint32_t* pp =
            (const uint32_t*)(sP + nl * ATT_BAND) + kt * 16;
#pragma unroll
        for (int j = 0; j < 8; ++j) a.u[j] = pp[aidx(j, half)];
#pragma unroll
        for (int e = 0; e < 16; ++e) {
            const int key = kstart + kt * 32 + e + 16 * half;
            const int kc  = key < 0 ? 0 : (key >= S ? S - 1 : key);
            b.h[e] = Vb[(size_t)kc * dm + dcol];   // P==0 there if masked
        }
        o = __builtin_amdgcn_wmma_f32_16x16x32_bf16(false, a.v, false, b.v,
                                                    (short)0, o, false, false);
    }
#pragma unroll
    for (int r = 0; r < 8; ++r) {
        const int row = r + 8 * half;
        const float val = o[r] / rowsum[row];
        Ab[(size_t)(q0 + row) * dm + dcol] = f2bf(val);
    }
}

// ---------------------------------------------------------------------------
// Residual add + LayerNorm: out = LN(xin + res) * g + b ; f32 + bf16 mirrors.
// One block per row (768 = 256 thr * 3 elems).
// ---------------------------------------------------------------------------
__global__ void add_ln_kernel(const float* __restrict__ xin,
                              const float* __restrict__ res,
                              const float* __restrict__ g,
                              const float* __restrict__ b,
                              float* __restrict__ xout,
                              unsigned short* __restrict__ xbf) {
    const int s = blockIdx.x, t = threadIdx.x;
    const size_t base = (size_t)s * 768;
    float v[3];
#pragma unroll
    for (int j = 0; j < 3; ++j) {
        const int i = t * 3 + j;
        float a = xin[base + i];
        if (res) a += res[base + i];
        v[j] = a;
    }
    __shared__ float red[256];
    float loc = v[0] + v[1] + v[2];
    red[t] = loc; __syncthreads();
    for (int off = 128; off; off >>= 1) {
        if (t < off) red[t] += red[t + off];
        __syncthreads();
    }
    const float mean = red[0] * (1.f / 768.f);
    __syncthreads();
    loc = 0.f;
#pragma unroll
    for (int j = 0; j < 3; ++j) { const float d = v[j] - mean; loc += d * d; }
    red[t] = loc; __syncthreads();
    for (int off = 128; off; off >>= 1) {
        if (t < off) red[t] += red[t + off];
        __syncthreads();
    }
    const float rs = rsqrtf(red[0] * (1.f / 768.f) + 1e-5f);
#pragma unroll
    for (int j = 0; j < 3; ++j) {
        const int i = t * 3 + j;
        const float y = (v[j] - mean) * rs * g[i] + b[i];
        xout[base + i] = y;
        xbf[base + i]  = f2bf(y);
    }
}

// Embedding gather: tmp[s,:] = emb_tok[tokens[s],:] + emb_pos[s,:]
__global__ void embed_kernel(const int* __restrict__ tokens,
                             const float* __restrict__ et,
                             const float* __restrict__ ep,
                             float* __restrict__ out) {
    const int s = blockIdx.x;
    const int tok = tokens[s];
    for (int i = threadIdx.x; i < 768; i += 256)
        out[(size_t)s * 768 + i] = et[(size_t)tok * 768 + i] + ep[(size_t)s * 768 + i];
}

// Weight convert+transpose: W[K,N] f32 -> Wt[N,K] bf16 (coalesced writes).
__global__ void conv_t_kernel(const float* __restrict__ W,
                              unsigned short* __restrict__ Wt, int K, int N) {
    const int idx = blockIdx.x * 256 + threadIdx.x;
    if (idx >= K * N) return;
    const int n = idx / K, k = idx - n * K;
    Wt[idx] = f2bf(W[(size_t)k * N + n]);
}

// ---------------------------------------------------------------------------
extern "C" void kernel_launch(void* const* d_in, const int* in_sizes, int n_in,
                              void* d_out, int out_size, void* d_ws, size_t ws_size,
                              hipStream_t stream) {
    (void)in_sizes; (void)n_in; (void)out_size; (void)ws_size;
    const int S = 4096, dm = 768, dff = 3072, L = 6, H = 12, w = 512;
    const float qscale = 0.125f;  // 1/sqrt(64)

    const int*   tokens  = (const int*)  d_in[0];
    const float* emb_tok = (const float*)d_in[1];
    const float* emb_pos = (const float*)d_in[2];
    const float* ln_e_s  = (const float*)d_in[3];
    const float* ln_e_b  = (const float*)d_in[4];
    const float* Wq = (const float*)d_in[5];  const float* bq = (const float*)d_in[6];
    const float* Wk = (const float*)d_in[7];  const float* bk = (const float*)d_in[8];
    const float* Wv = (const float*)d_in[9];  const float* bv = (const float*)d_in[10];
    const float* Wo = (const float*)d_in[11]; const float* bo = (const float*)d_in[12];
    const float* ln1s = (const float*)d_in[13]; const float* ln1b = (const float*)d_in[14];
    const float* W1f = (const float*)d_in[15]; const float* b1f = (const float*)d_in[16];
    const float* W2f = (const float*)d_in[17]; const float* b2f = (const float*)d_in[18];
    const float* ln2s = (const float*)d_in[19]; const float* ln2b = (const float*)d_in[20];

    char* ws = (char*)d_ws;
    size_t off = 0;
    auto alloc = [&](size_t bytes) -> void* {
        void* p = ws + off;
        off += (bytes + 255) & ~(size_t)255;
        return p;
    };
    float*          x    = (float*)alloc((size_t)S * dm * 4);
    float*          tmp  = (float*)alloc((size_t)S * dm * 4);
    unsigned short* xbf  = (unsigned short*)alloc((size_t)S * dm * 2);
    unsigned short* qbf  = (unsigned short*)alloc((size_t)S * dm * 2);
    unsigned short* kbf  = (unsigned short*)alloc((size_t)S * dm * 2);
    unsigned short* vbf  = (unsigned short*)alloc((size_t)S * dm * 2);
    unsigned short* abf  = (unsigned short*)alloc((size_t)S * dm * 2);
    unsigned short* hbf  = (unsigned short*)alloc((size_t)S * dff * 2);
    unsigned short* wqT  = (unsigned short*)alloc((size_t)dm * dm * 2);
    unsigned short* wkT  = (unsigned short*)alloc((size_t)dm * dm * 2);
    unsigned short* wvT  = (unsigned short*)alloc((size_t)dm * dm * 2);
    unsigned short* woT  = (unsigned short*)alloc((size_t)dm * dm * 2);
    unsigned short* w1T  = (unsigned short*)alloc((size_t)dm * dff * 2);
    unsigned short* w2T  = (unsigned short*)alloc((size_t)dff * dm * 2);

    auto gemm = [&](const unsigned short* A, const unsigned short* Bt,
                    const float* bias, float* Cf, unsigned short* Cb,
                    int M, int N, int K, float alpha, int gelu) {
        const int tiles = (M / 16) * (N / 64);   // 16x64 strip per wave
        gemm_bf16_wmma<<<(tiles + 7) / 8, 256, 0, stream>>>(
            A, Bt, bias, Cf, Cb, M, N, K, alpha, gelu);
    };
    auto convt = [&](const float* W, unsigned short* Wt, int K, int N) {
        const int tot = K * N;
        conv_t_kernel<<<(tot + 255) / 256, 256, 0, stream>>>(W, Wt, K, N);
    };

    // Embedding + encoder LN
    embed_kernel<<<S, 256, 0, stream>>>(tokens, emb_tok, emb_pos, tmp);
    add_ln_kernel<<<S, 256, 0, stream>>>(tmp, nullptr, ln_e_s, ln_e_b, x, xbf);

    for (int l = 0; l < L; ++l) {
        convt(Wq  + (size_t)l * dm * dm,  wqT, dm, dm);
        convt(Wk  + (size_t)l * dm * dm,  wkT, dm, dm);
        convt(Wv  + (size_t)l * dm * dm,  wvT, dm, dm);
        convt(Wo  + (size_t)l * dm * dm,  woT, dm, dm);
        convt(W1f + (size_t)l * dm * dff, w1T, dm, dff);
        convt(W2f + (size_t)l * dff * dm, w2T, dff, dm);

        gemm(xbf, wqT, bq + l * dm, nullptr, qbf, S, dm, dm, qscale, 0);
        gemm(xbf, wkT, bk + l * dm, nullptr, kbf, S, dm, dm, 1.f, 0);
        gemm(xbf, wvT, bv + l * dm, nullptr, vbf, S, dm, dm, 1.f, 0);

        attn_window_wmma<<<dim3(S / 16, H), 128, 0, stream>>>(
            qbf, kbf, vbf, abf, S, dm, w);

        gemm(abf, woT, bo + l * dm, tmp, nullptr, S, dm, dm, 1.f, 0);
        add_ln_kernel<<<S, 256, 0, stream>>>(x, tmp, ln1s + l * dm, ln1b + l * dm, x, xbf);

        gemm(xbf, w1T, b1f + l * dff, nullptr, hbf, S, dff, dm, 1.f, 1);
        gemm(hbf, w2T, b2f + l * dm, tmp, nullptr, S, dm, dff, 1.f, 0);
        add_ln_kernel<<<S, 256, 0, stream>>>(x, tmp, ln2s + l * dm, ln2b + l * dm, x, xbf);
    }

    hipMemcpyAsync(d_out, x, (size_t)S * dm * sizeof(float),
                   hipMemcpyDeviceToDevice, stream);
}
